// myVMLSTM_455266534065
// MI455X (gfx1250) — compile-verified, matrix operand order checked by reading, not run on hardware
//
#include <hip/hip_runtime.h>

// ---------------------------------------------------------------------------
// VMLSTM for MI455X (gfx1250, wave32, WMMA + TDM).
//   T=512, B=64, I=H=512, RW=RU=64.
//   prep_misc : ddx = dia_x - Dx, ddh = dia_h - Dh, bias = b_x + b_h   [2048 f32]
//   prep_uwmma: U_h / U_x  -> WMMA-ready bf16 B-operand images
//   prep_bwmma: [W_h^T ; W_x^T] (K=128, N=2048) -> WMMA-ready bf16 image (512 KB)
//   prep_P    : P = x @ U_x  (bf16 WMMA, fp32 accum, dual accumulation chains)
//   vmlstm    : 4 persistent WGs (batch blocks of 16 = WMMA M). Per step:
//               q = h @ U_h (WMMA), z = [q|P_t] @ Bcat (WMMA), gates in VALU.
//               P_t tile staged by the Tensor Data Mover (tensor_load_to_lds,
//               pad_enable packs rows at col 64 of the [q|P] LDS buffer),
//               software-pipelined one step ahead, synced by s_wait_tensorcnt.
//               Recurrence is independent per batch -> no inter-WG sync.
// ---------------------------------------------------------------------------

typedef __attribute__((ext_vector_type(16))) __bf16 v16bf;
typedef __attribute__((ext_vector_type(8)))  float  v8f;
typedef __attribute__((ext_vector_type(4)))  unsigned int u32x4;
typedef __attribute__((ext_vector_type(8)))  int i32x8;
typedef __attribute__((ext_vector_type(4)))  int i32x4;

#define T_  512
#define BB  64
#define HH  512
#define RR  64
#define G4H 2048
#define BC  16        // batch rows per workgroup (== WMMA M)
#define ZLD 2052      // padded z row stride (floats): keeps row halves on distinct banks

// fp32 -> bf16, round-to-nearest-even
__device__ __forceinline__ unsigned short f2bf(float f) {
    unsigned int u = __float_as_uint(f);
    u += 0x7FFFu + ((u >> 16) & 1u);
    return (unsigned short)(u >> 16);
}
__device__ __forceinline__ float bf2f(unsigned short s) {
    return __uint_as_float(((unsigned int)s) << 16);
}

// two 16B loads -> one 16x bf16 WMMA operand fragment
__device__ __forceinline__ v16bf load2(const unsigned short* p0, const unsigned short* p1) {
    union { v16bf v; u32x4 u[2]; } r;
    r.u[0] = *(const u32x4*)p0;
    r.u[1] = *(const u32x4*)p1;
    return r.v;
}

__device__ __forceinline__ v8f wmma_bf16(v16bf a, v16bf b, v8f c) {
    return __builtin_amdgcn_wmma_f32_16x16x32_bf16(false, a, false, b, (short)0, c,
                                                   false, false);
}

__device__ __forceinline__ float sigmoid_(float v) {
    return __builtin_amdgcn_rcpf(1.0f + __expf(-v));
}
__device__ __forceinline__ float tanh_(float v) {
    return 1.0f - 2.0f * __builtin_amdgcn_rcpf(1.0f + __expf(2.0f * v));
}

// --------------------------------------------------------------------------
// TDM: load one [16 x 64] bf16 P tile into LDS with 128B row padding, so each
// 128B row lands at column 64 of the 16x128 [q|P] LDS buffer.
// D# bitfields per CDNA5 ISA 8.3/8.4:
//   g0: count=1 | lds_addr | global_addr[56:0] | type=2
//   g1: data_size=1(2B), pad_enable, pad_interval=4 (16x8B=128B),
//       pad_amount=31 (32 dwords=128B), tensor_dim0=64, tensor_dim1=0x100000,
//       tile_dim0=64, tile_dim1=16, tensor_dim0_stride=64
// This toolchain's builtin takes 6 args: (g0, g1, g2, g3, g_extra, cpol).
// --------------------------------------------------------------------------
__device__ __forceinline__ void tdm_load_P(const unsigned short* Pbf, int t, int b0,
                                           unsigned ldsAddr) {
    unsigned long long ga =
        (unsigned long long)(const void*)(Pbf + ((size_t)t * BB + b0) * RR);
    u32x4 g0 = { 1u,                                    // count=1, user mode
                 ldsAddr,                               // LDS byte address
                 (unsigned)(ga & 0xffffffffu),          // global_addr[31:0]
                 (unsigned)((ga >> 32) & 0x1ffffffu) |  // global_addr[56:32]
                     (2u << 30) };                      // type=2 ("image")
    i32x8 g1 = { (int)0x3F110000u,        // pad_amt=31|pad_int=4|pad_en|data_size=1
                 (int)(64u << 16),        // tensor_dim0[15:0]=64 @ bits[31:16]
                 0,                       // tensor_dim0[31:16]=0, tensor_dim1[15:0]=0
                 (int)((64u << 16) | 16u),// tile_dim0=64 | tensor_dim1[31:16] (dim1=1M)
                 16,                      // tile_dim1=16, tile_dim2=0
                 64,                      // tensor_dim0_stride[31:0]=64
                 0, 0 };
    i32x4 gz4 = { 0, 0, 0, 0 };
    i32x8 gz8 = { 0, 0, 0, 0, 0, 0, 0, 0 };
    __builtin_amdgcn_tensor_load_to_lds(g0, g1, gz4, gz4, gz8, 0);
}

// --------------------------------------------------------------------------
// ddx[n]=dia_x[i]-sum_r U_x[i,r]W_x[n,r]; ddh likewise; bias=b_x+b_h.  n=g*512+i
// --------------------------------------------------------------------------
__global__ void prep_misc(const float* __restrict__ U_x, const float* __restrict__ U_h,
                          const float* __restrict__ W_x, const float* __restrict__ W_h,
                          const float* __restrict__ b_x, const float* __restrict__ b_h,
                          const float* __restrict__ dia_x, const float* __restrict__ dia_h,
                          float* __restrict__ ddx, float* __restrict__ ddh,
                          float* __restrict__ bias) {
    int n = blockIdx.x * blockDim.x + threadIdx.x;
    if (n >= G4H) return;
    int i = n & (HH - 1);
    float sx = 0.f, sh = 0.f;
    #pragma unroll 8
    for (int r = 0; r < RR; ++r) {
        sx += U_x[i * RR + r] * W_x[n * RR + r];
        sh += U_h[i * RR + r] * W_h[n * RR + r];
    }
    ddx[n]  = dia_x[i] - sx;
    ddh[n]  = dia_h[i] - sh;
    bias[n] = b_x[n] + b_h[n];
}

// --------------------------------------------------------------------------
// WMMA-ready image of a [K=512, N=64] B-matrix (U_h or U_x):
//   Uw[((nt*16+ks)*32 + lane)*16 + j] = bf16(U[k,n]),  per 16-bit B VGPR layout
// --------------------------------------------------------------------------
__global__ void prep_uwmma(const float* __restrict__ U, unsigned short* __restrict__ Uw) {
    int idx = blockIdx.x * blockDim.x + threadIdx.x;          // 32768
    int j    = idx & 15;
    int lane = (idx >> 4) & 31;
    int ks   = (idx >> 9) & 15;
    int nt   = idx >> 13;                                     // 0..3
    int n = (nt << 4) + (lane & 15);
    int v = j >> 1;
    int k = ks * 32 + ((v >> 2) << 4) + ((lane >> 4) << 3) + ((v & 3) << 1) + (j & 1);
    Uw[idx] = f2bf(U[k * RR + n]);
}

// --------------------------------------------------------------------------
// WMMA-ready image of Bcat [K=128, N=2048]: rows 0..63 = W_h^T, 64..127 = W_x^T
// --------------------------------------------------------------------------
__global__ void prep_bwmma(const float* __restrict__ W_h, const float* __restrict__ W_x,
                           unsigned short* __restrict__ Bw) {
    int idx = blockIdx.x * blockDim.x + threadIdx.x;          // 262144
    int j    = idx & 15;
    int lane = (idx >> 4) & 31;
    int ks   = (idx >> 9) & 3;
    int nt   = idx >> 11;                                     // 0..127
    int n = (nt << 4) + (lane & 15);
    int v = j >> 1;
    int k = ks * 32 + ((v >> 2) << 4) + ((lane >> 4) << 3) + ((v & 3) << 1) + (j & 1);
    float val = (k < RR) ? W_h[n * RR + k] : W_x[n * RR + (k - RR)];
    Bw[idx] = f2bf(val);
}

// --------------------------------------------------------------------------
// P = x @ U_x : [32768, 64] bf16.  One block per 16-row M-tile, 4 waves,
// wave w = N-tile w.  Two interleaved accumulation chains hide the
// WMMA->WMMA RAW hazard (1 NOP + 4 co-exec for bf16).
// --------------------------------------------------------------------------
__global__ __launch_bounds__(128) void prep_P(const float* __restrict__ x,
                                              const unsigned short* __restrict__ Uxw,
                                              unsigned short* __restrict__ Pbf) {
    __shared__ unsigned short sx[16 * HH];                    // 16 KB bf16 tile
    const int tid = threadIdx.x;
    const int ri0 = blockIdx.x * 16;
    const float* xs = x + (size_t)ri0 * HH;
    for (int e = tid; e < 16 * HH; e += 128) sx[e] = f2bf(xs[e]);
    __syncthreads();

    const int lane = tid & 31, wv = tid >> 5;
    const int lh = lane >> 4, ln = lane & 15;
    v8f acc0 = {}, acc1 = {};
    #pragma unroll
    for (int ks2 = 0; ks2 < 8; ++ks2) {
        const int ksA = 2 * ks2, ksB = 2 * ks2 + 1;
        const unsigned short* apA = sx + ln * HH + ksA * 32 + lh * 8;
        const unsigned short* apB = sx + ln * HH + ksB * 32 + lh * 8;
        v16bf aA = load2(apA, apA + 16);
        v16bf aB = load2(apB, apB + 16);
        const unsigned short* bpA = Uxw + (((wv * 16 + ksA) * 32) + lane) * 16;
        const unsigned short* bpB = Uxw + (((wv * 16 + ksB) * 32) + lane) * 16;
        v16bf bA = load2(bpA, bpA + 8);
        v16bf bB = load2(bpB, bpB + 8);
        acc0 = wmma_bf16(aA, bA, acc0);
        acc1 = wmma_bf16(aB, bB, acc1);
    }
    v8f acc = acc0 + acc1;
    #pragma unroll
    for (int r = 0; r < 8; ++r)
        Pbf[(size_t)(ri0 + r + 8 * lh) * RR + wv * 16 + ln] = f2bf(acc[r]);
}

// --------------------------------------------------------------------------
// Persistent recurrent kernel: grid = 4 (batch blocks of 16), 512 threads.
// LDS ~148 KB (safe even if CU-mode halves the 320 KB WGP pool).
// --------------------------------------------------------------------------
__global__ __launch_bounds__(512) void vmlstm_scan(
        const float* __restrict__ x, const float* __restrict__ h0,
        const float* __restrict__ c0, const unsigned short* __restrict__ Pbf,
        const unsigned short* __restrict__ Uw, const unsigned short* __restrict__ Bw,
        const float* __restrict__ ddx, const float* __restrict__ ddh,
        const float* __restrict__ bias, float* __restrict__ out) {
    extern __shared__ char smem[];
    float*          sZ   = (float*)smem;                        // 16*ZLD f32
    unsigned short* sHbf = (unsigned short*)(sZ + 16 * ZLD);    // 16*512 bf16
    unsigned short* sA2  = sHbf + BC * HH;                      // 16*128 bf16 [q|P]

    const int tid  = threadIdx.x;
    const int lane = tid & 31, wv = tid >> 5;
    const int lh = lane >> 4, ln = lane & 15;
    const int b0 = blockIdx.x * BC;
    // LDS byte address of the P region (col 64 of row 0 of sA2); low 32 bits of
    // a flat shared pointer are the LDS offset.
    const unsigned ldsA2P =
        (unsigned)(unsigned long long)(const void*)sA2 + 64u * 2u;

    // state ownership: thread = (batch row sb, 16 contiguous i starting at i0)
    const int sb = tid >> 5;
    const int i0 = (tid & 31) * 16;
    float c[16];
    #pragma unroll
    for (int ii = 0; ii < 16; ++ii) {
        c[ii] = c0[(b0 + sb) * HH + i0 + ii];
        sHbf[sb * HH + i0 + ii] = f2bf(h0[(b0 + sb) * HH + i0 + ii]);
    }
    if (wv == 4) tdm_load_P(Pbf, 0, b0, ldsA2P);   // stage P tile for t=0
    __syncthreads();

    for (int t = 0; t < T_; ++t) {
        // ---- phase 1: waves 0..3 -> q = h @ U_h (two accumulation chains) ----
        if (wv < 4) {
            v8f q0 = {}, q1 = {};
            #pragma unroll
            for (int ks2 = 0; ks2 < 8; ++ks2) {
                const int ksA = 2 * ks2, ksB = 2 * ks2 + 1;
                const unsigned short* apA = sHbf + ln * HH + ksA * 32 + lh * 8;
                const unsigned short* apB = sHbf + ln * HH + ksB * 32 + lh * 8;
                v16bf aA = load2(apA, apA + 16);
                v16bf aB = load2(apB, apB + 16);
                const unsigned short* bpA = Uw + (((wv * 16 + ksA) * 32) + lane) * 16;
                const unsigned short* bpB = Uw + (((wv * 16 + ksB) * 32) + lane) * 16;
                v16bf bA = load2(bpA, bpA + 8);
                v16bf bB = load2(bpB, bpB + 8);
                q0 = wmma_bf16(aA, bA, q0);
                q1 = wmma_bf16(aB, bB, q1);
            }
            v8f q = q0 + q1;
            #pragma unroll
            for (int r = 0; r < 8; ++r)
                sA2[(r + 8 * lh) * 128 + wv * 16 + ln] = f2bf(q[r]);
        }
        if (wv == 4) __builtin_amdgcn_s_wait_tensorcnt(0);  // P tile (this step) done
        __syncthreads();

        // ---- phase 2: z[16,2048] = [q|P] @ Bcat ; 8 N-tiles per wave ----
        {
            v8f acc[8] = {};
            #pragma unroll
            for (int ks = 0; ks < 4; ++ks) {
                const unsigned short* ap = sA2 + ln * 128 + ks * 32 + lh * 8;
                v16bf a = load2(ap, ap + 16);
                #pragma unroll
                for (int j = 0; j < 8; ++j) {
                    int nt = wv * 8 + j;
                    const unsigned short* bp = Bw + ((nt * 4 + ks) * 32 + lane) * 16;
                    v16bf b = load2(bp, bp + 8);
                    acc[j] = wmma_bf16(a, b, acc[j]);
                }
            }
            #pragma unroll
            for (int j = 0; j < 8; ++j) {
                int nt = wv * 8 + j;
                #pragma unroll
                for (int r = 0; r < 8; ++r)
                    sZ[(r + 8 * lh) * ZLD + nt * 16 + ln] = acc[j][r];
            }
        }
        __syncthreads();

        // pipelined TDM: stage P tile for t+1 (P region free after phase-2 barrier)
        if (wv == 4 && t + 1 < T_) tdm_load_P(Pbf, t + 1, b0, ldsA2P);

        // ---- phase 3: gates + state update ----
        {
            const float* xrow = x + ((size_t)t * BB + b0 + sb) * HH + i0;
            float*       orow = out + ((size_t)t * BB + b0 + sb) * HH + i0;
            float*       zrow = sZ + sb * ZLD;
            const bool last = (t == T_ - 1);
            #pragma unroll
            for (int ii = 0; ii < 16; ++ii) {
                const int n0 = i0 + ii;
                const float xv = xrow[ii];
                const float hv = bf2f(sHbf[sb * HH + n0]);
                float gi = zrow[n0]        + xv * ddx[n0]        + hv * ddh[n0]        + bias[n0];
                float gf = zrow[512 + n0]  + xv * ddx[512 + n0]  + hv * ddh[512 + n0]  + bias[512 + n0];
                float go = zrow[1024 + n0] + xv * ddx[1024 + n0] + hv * ddh[1024 + n0] + bias[1024 + n0];
                float gn = zrow[1536 + n0] + xv * ddx[1536 + n0] + hv * ddh[1536 + n0] + bias[1536 + n0];
                float ig = sigmoid_(gi), fg = sigmoid_(gf), og = sigmoid_(go);
                float ng = tanh_(gn);
                c[ii] = fg * c[ii] + ig * ng;
                float hnew = og * tanh_(c[ii]);
                orow[ii] = hnew;
                sHbf[sb * HH + n0] = f2bf(hnew);
                if (last)
                    out[(size_t)T_ * BB * HH + (b0 + sb) * HH + n0] = hnew;
            }
        }
        __syncthreads();
    }

    const size_t cbase = (size_t)T_ * BB * HH + (size_t)BB * HH;
    #pragma unroll
    for (int ii = 0; ii < 16; ++ii)
        out[cbase + (b0 + sb) * HH + i0 + ii] = c[ii];
}

// --------------------------------------------------------------------------
extern "C" void kernel_launch(void* const* d_in, const int* in_sizes, int n_in,
                              void* d_out, int out_size, void* d_ws, size_t ws_size,
                              hipStream_t stream) {
    (void)in_sizes; (void)n_in; (void)out_size; (void)ws_size;
    const float* x     = (const float*)d_in[0];
    const float* h0    = (const float*)d_in[1];
    const float* c0    = (const float*)d_in[2];
    const float* U_x   = (const float*)d_in[3];
    const float* U_h   = (const float*)d_in[4];
    const float* W_x   = (const float*)d_in[5];
    const float* W_h   = (const float*)d_in[6];
    const float* b_x   = (const float*)d_in[7];
    const float* b_h   = (const float*)d_in[8];
    const float* dia_x = (const float*)d_in[9];
    const float* dia_h = (const float*)d_in[10];
    float* out = (float*)d_out;

    char* ws = (char*)d_ws;
    float* ddx  = (float*)ws;  ws += G4H * sizeof(float);
    float* ddh  = (float*)ws;  ws += G4H * sizeof(float);
    float* bias = (float*)ws;  ws += G4H * sizeof(float);
    unsigned short* Uw  = (unsigned short*)ws; ws += 4   * 16 * 32 * 16 * 2; //  64 KB
    unsigned short* Uxw = (unsigned short*)ws; ws += 4   * 16 * 32 * 16 * 2; //  64 KB
    unsigned short* Bw  = (unsigned short*)ws; ws += 128 * 4  * 32 * 16 * 2; // 512 KB
    unsigned short* Pbf = (unsigned short*)ws;                               //   4 MB

    prep_misc <<<(G4H + 255) / 256, 256, 0, stream>>>(U_x, U_h, W_x, W_h, b_x, b_h,
                                                      dia_x, dia_h, ddx, ddh, bias);
    prep_uwmma<<<64, 512, 0, stream>>>(U_h, Uw);
    prep_uwmma<<<64, 512, 0, stream>>>(U_x, Uxw);
    prep_bwmma<<<512, 512, 0, stream>>>(W_h, W_x, Bw);
    prep_P    <<<(T_ * BB) / 16, 128, 0, stream>>>(x, Uxw, Pbf);

    const size_t smem = 16 * ZLD * sizeof(float)           // z
                      + BC * HH * sizeof(unsigned short)   // h (bf16)
                      + BC * 128 * sizeof(unsigned short); // [q|P]
    vmlstm_scan<<<BB / BC, 512, smem, stream>>>(x, h0, c0, Pbf, Uw, Bw,
                                                ddx, ddh, bias, out);
}
